// LSTMAutoencoder_24481313587455
// MI455X (gfx1250) — matrix operand order repaired and checked by reading, not compiled, not run on hardware
//
#include <hip/hip_runtime.h>
#include <hip/hip_bf16.h>

// ---------------------------------------------------------------------------
// LSTM autoencoder for MI455X (gfx1250, wave32).
//  - bf16 V_WMMA_F32_16X16X32_BF16 for all matmuls (8x K-depth vs fp32 WMMA)
//  - input projections as parallel WMMA GEMMs into workspace xg buffers
//  - per-layer sequential scan: block owns 16 batch rows, Whh pre-packed into
//    VGPR B-fragments for the whole T loop, h in LDS, c in registers.
//  - Tensor Data Mover (tensor_load_to_lds) double-buffers the next step's
//    16x4H xg gate tile into LDS while the current step computes; consumers
//    gate on s_wait_tensorcnt. Falls back to direct global loads + prefetch
//    if the TDM builtin is unavailable.
// ---------------------------------------------------------------------------

typedef float        v8f   __attribute__((ext_vector_type(8)));
typedef __bf16       v16bf __attribute__((ext_vector_type(16)));
typedef __bf16       v8bf  __attribute__((ext_vector_type(8)));
typedef unsigned int v4u   __attribute__((ext_vector_type(4)));
typedef int          v8i   __attribute__((ext_vector_type(8)));
typedef int          v4i   __attribute__((ext_vector_type(4)));

#define DEVFN __device__ __forceinline__

#if __has_builtin(__builtin_amdgcn_tensor_load_to_lds) && \
    __has_builtin(__builtin_amdgcn_s_wait_tensorcnt)
#define USE_TDM 1
#else
#define USE_TDM 0
#endif

DEVFN v8f wmma_bf16(v16bf a, v16bf b, v8f c) {
  // (neg_a, A, neg_b, B, c_mod, C, reuse_a, reuse_b)
  return __builtin_amdgcn_wmma_f32_16x16x32_bf16(false, a, false, b,
                                                 (short)0, c, false, false);
}

// A fragment: row-major source, lane holds row (lane&15),
// K elems {kb..kb+7} and {kb+16..kb+23}, kb = (lane>>4)*8. p -> elem kb.
DEVFN v16bf load_a_frag(const __bf16* p) {
  union { v16bf v; v8bf h[2]; } u;
  u.h[0] = *(const v8bf*)(p);
  u.h[1] = *(const v8bf*)(p + 16);
  return u.v;
}

// B fragment: lane holds column (lane&15) of W^T == row of W[N,K];
// K elems {kb..kb+7} and {kb+16..kb+23}, kb = (lane>>4)*8. p -> elem kb.
DEVFN v16bf load_b_frag(const __bf16* p) {
  union { v16bf v; v8bf h[2]; } u;
  u.h[0] = *(const v8bf*)(p);
  u.h[1] = *(const v8bf*)(p + 16);
  return u.v;
}

DEVFN float sigmf(float x) { return 1.0f / (1.0f + __expf(-x)); }

#if USE_TDM
// Issue a TDM load of a 2-D fp32 tile [16 rows x dim0 cols] into LDS.
// Rows are strided by row_stride_elems in global memory; LDS dest contiguous.
// D# packing per CDNA5 ISA.
DEVFN void tdm_load_tile_f32(unsigned int lds_off, const float* gptr, int dim0,
                             long long row_stride_elems) {
  unsigned long long ga = (unsigned long long)(size_t)gptr;
  v4u g0;
  g0[0] = 1u;                                  // count=1 (valid user D#)
  g0[1] = lds_off;                             // lds_addr (bytes)
  g0[2] = (unsigned int)ga;                    // global_addr[31:0]
  g0[3] = (unsigned int)(ga >> 32) | (2u << 30);  // global_addr[56:32]|type=2
  unsigned long long rs = (unsigned long long)row_stride_elems;
  v8i g1;
  g1[0] = 0x00020000;                          // data_size=2 (4 bytes)
  g1[1] = (int)((dim0 & 0xFFFF) << 16);        // tensor_dim0[15:0]
  g1[2] = (int)(((unsigned)dim0 >> 16) | (16u << 16));  // dim0 hi | dim1=16
  g1[3] = (int)((dim0 & 0xFFFF) << 16);        // dim1 hi=0 | tile_dim0=dim0
  g1[4] = 16;                                  // tile_dim1=16, tile_dim2=0
  g1[5] = (int)(unsigned int)(rs & 0xFFFFFFFFull);      // dim0_stride lo32
  g1[6] = (int)(unsigned int)((rs >> 32) & 0xFFFFull);  // dim0_stride hi16
  g1[7] = 0;                                   // dim1_stride hi = 0
  v4i z4 = {0, 0, 0, 0};
#if defined(__clang_major__) && (__clang_major__ >= 23)
  v8i z8 = {0, 0, 0, 0, 0, 0, 0, 0};
  __builtin_amdgcn_tensor_load_to_lds(g0, g1, z4, z4, z8, 0);
#else
  __builtin_amdgcn_tensor_load_to_lds(g0, g1, z4, z4, 0);
#endif
}
#endif

// ---------------------------------------------------------------- converters
__global__ void k_f32_to_bf16(const float* __restrict__ s,
                              __bf16* __restrict__ d, long long n) {
  long long i = (long long)blockIdx.x * blockDim.x + threadIdx.x;
  if (i < n) d[i] = (__bf16)s[i];
}

// fp32 [N,K] -> bf16 [N,KP], zero-padded K..KP-1
__global__ void k_cvt_pad(const float* __restrict__ s, __bf16* __restrict__ d,
                          int N, int K, int KP) {
  int i = blockIdx.x * blockDim.x + threadIdx.x;
  if (i < N * KP) {
    int n = i / KP, k = i % KP;
    d[i] = (k < K) ? (__bf16)s[n * K + k] : (__bf16)0.0f;
  }
}

// ---------------------------------------------------- input-projection GEMM
// C[M,N] = A[M,K](bf16) @ W[N,K](bf16)^T + bih + bhh  (fp32 out)
// grid.x = M/64 (4 waves x 16 rows), grid.y = N/64 (4 N-tiles per wave)
__global__ __launch_bounds__(128) void k_gemm_xg(
    const __bf16* __restrict__ A, const __bf16* __restrict__ W,
    const float* __restrict__ bih, const float* __restrict__ bhh,
    float* __restrict__ C, int K, int N) {
  const int lane = threadIdx.x & 31, wave = threadIdx.x >> 5;
  const int m0 = (blockIdx.x * 4 + wave) * 16;
  const int n0 = blockIdx.y * 64;
  const int kbA = (lane >> 4) * 8;
  const int kbB = (lane >> 4) * 8;
  const __bf16* Abase = A + (size_t)(m0 + (lane & 15)) * K;

  v8f acc[4];
  for (int nt = 0; nt < 4; ++nt)
    for (int r = 0; r < 8; ++r) acc[nt][r] = 0.0f;

  for (int kc = 0; kc < K; kc += 32) {
    v16bf a = load_a_frag(Abase + kc + kbA);
    for (int nt = 0; nt < 4; ++nt) {
      int ncol = n0 + nt * 16 + (lane & 15);
      v16bf b = load_b_frag(W + (size_t)ncol * K + kc + kbB);
      acc[nt] = wmma_bf16(a, b, acc[nt]);
    }
  }
  for (int nt = 0; nt < 4; ++nt) {
    int ncol = n0 + nt * 16 + (lane & 15);
    float bias = bih[ncol] + bhh[ncol];
    for (int r = 0; r < 8; ++r) {
      int row = m0 + r + ((lane >> 4) * 8);
      C[(size_t)row * N + ncol] = acc[nt][r] + bias;
    }
  }
}

// ----------------------------------------------- d1 projection (latent, K=16)
// xg_d1[b, 0:256] = latent[b] @ Wih_d1^T + bih + bhh ; latent = out_e2[b,T-1,:]
__global__ void k_latent_xg(const __bf16* __restrict__ out_e2,
                            const float* __restrict__ Wih,
                            const float* __restrict__ bih,
                            const float* __restrict__ bhh,
                            float* __restrict__ xg, int T) {
  int i = blockIdx.x * blockDim.x + threadIdx.x;  // B*256
  int b = i >> 8, n = i & 255;
  const __bf16* lat = out_e2 + ((size_t)b * T + (T - 1)) * 16;
  float s = bih[n] + bhh[n];
  for (int k = 0; k < 16; ++k) s += (float)lat[k] * Wih[n * 16 + k];
  xg[i] = s;
}

// ------------------------------------------------------------ recurrent scan
// Block = 16 batch rows. Per step: g = h @ Whh^T (WMMA, Whh resident in VGPR
// B-fragments), gates += xg_t (TDM-staged in LDS), elementwise c/h update,
// h -> LDS, out -> global. TDM double-buffers xg tile t+1 during step t.
template <int H, int NWAVES, bool RELU, typename OutT>
__global__ __launch_bounds__(NWAVES * 32) void k_lstm_scan(
    const float* __restrict__ xg, const __bf16* __restrict__ Whh /*[4H,KPAD]*/,
    OutT* __restrict__ out, int T, long long xg_b_stride,
    long long xg_t_stride) {
  constexpr int NT = (4 * H) / 16;      // N-tiles of gates
  constexpr int NTPW = NT / NWAVES;     // N-tiles per wave
  constexpr int KC = (H + 31) / 32;     // 32-wide K chunks
  constexpr int KPAD = KC * 32;
  constexpr int NTHR = NWAVES * 32;
  constexpr int EPT = (16 * H) / NTHR;  // c/h elements per thread

  __shared__ alignas(32) __bf16 hbuf[16 * KPAD];
  __shared__ float gbuf[16 * 4 * H];
#if USE_TDM
  __shared__ alignas(16) float xbuf[2][16 * 4 * H];  // TDM double buffer
#endif

  const int tid = threadIdx.x;
  const int lane = tid & 31, wave = tid >> 5;
  const int b0 = blockIdx.x * 16;
  const int hrow = lane & 15;
  const int kbA = (lane >> 4) * 8;
  const int kbB = (lane >> 4) * 8;

  // Whh B-fragments resident in VGPRs for the whole scan
  v16bf breg[NTPW][KC];
  for (int nt = 0; nt < NTPW; ++nt) {
    int ncol = (wave * NTPW + nt) * 16 + hrow;
    for (int kc = 0; kc < KC; ++kc)
      breg[nt][kc] = load_b_frag(Whh + (size_t)ncol * KPAD + kc * 32 + kbB);
  }

  for (int i = tid; i < 16 * KPAD; i += NTHR) hbuf[i] = (__bf16)0.0f;
  float creg[EPT];
  for (int e = 0; e < EPT; ++e) creg[e] = 0.0f;

#if USE_TDM
  if (wave == 0) {  // stage tile for t=0 (wave-uniform branch; TDM ignores EXEC)
    unsigned int lds0 = (unsigned int)(size_t)(&xbuf[0][0]);
    tdm_load_tile_f32(lds0, xg + (size_t)b0 * xg_b_stride, 4 * H, xg_b_stride);
  }
#endif
  __syncthreads();

  for (int t = 0; t < T; ++t) {
#if USE_TDM
    if (wave == 0 && t + 1 < T) {  // stage tile t+1 while computing step t
      unsigned int ldsn = (unsigned int)(size_t)(&xbuf[(t + 1) & 1][0]);
      tdm_load_tile_f32(ldsn,
                        xg + (size_t)b0 * xg_b_stride +
                            (size_t)(t + 1) * xg_t_stride,
                        4 * H, xg_b_stride);
    }
#endif
    // A fragments (h) from LDS — shared across this wave's N-tiles
    v16bf afr[KC];
    for (int kc = 0; kc < KC; ++kc)
      afr[kc] = load_a_frag(&hbuf[hrow * KPAD + kc * 32 + kbA]);

    for (int nt = 0; nt < NTPW; ++nt) {
      v8f acc;
      for (int r = 0; r < 8; ++r) acc[r] = 0.0f;
      for (int kc = 0; kc < KC; ++kc)
        acc = wmma_bf16(afr[kc], breg[nt][kc], acc);
      int ncol = (wave * NTPW + nt) * 16 + hrow;
      for (int r = 0; r < 8; ++r)
        gbuf[(r + ((lane >> 4) * 8)) * 4 * H + ncol] = acc[r];
    }
#if USE_TDM
    // Older TDM op (tile t) complete; newer (tile t+1) may remain in flight.
    if (t + 1 < T)
      __builtin_amdgcn_s_wait_tensorcnt(1);
    else
      __builtin_amdgcn_s_wait_tensorcnt(0);
#endif
    __syncthreads();

#if USE_TDM
    const float* xr0 = &xbuf[t & 1][0];
#endif
    for (int e = 0; e < EPT; ++e) {
      int idx = e * NTHR + tid;
      int row = idx / H, j = idx % H;
#if USE_TDM
      const float* xr = xr0 + row * 4 * H;
#else
      const float* xr =
          xg + (size_t)(b0 + row) * xg_b_stride + (size_t)t * xg_t_stride;
#endif
      float gi = gbuf[row * 4 * H + j] + xr[j];
      float gf = gbuf[row * 4 * H + H + j] + xr[H + j];
      float gc = gbuf[row * 4 * H + 2 * H + j] + xr[2 * H + j];
      float go = gbuf[row * 4 * H + 3 * H + j] + xr[3 * H + j];
      float c = sigmf(gf) * creg[e] + sigmf(gi) * tanhf(gc);
      creg[e] = c;
      float hv = sigmf(go) * tanhf(c);
      if (RELU) hv = fmaxf(hv, 0.0f);
      hbuf[row * KPAD + j] = (__bf16)hv;
      out[((size_t)(b0 + row) * T + t) * H + j] = (OutT)hv;
#if !USE_TDM
      if (t + 1 < T) {  // prefetch next step's gate row (global_prefetch_b8)
        const float* xn = xr + xg_t_stride;
        __builtin_prefetch(xn + j, 0, 1);
        __builtin_prefetch(xn + 2 * H + j, 0, 1);
      }
#endif
    }
    __syncthreads();
  }
}

// ---------------------------------------------------------------------------
static constexpr int B_ = 256, T_ = 512, IN_ = 128, HID_ = 64, LAT_ = 16;
static constexpr long long BT_ = (long long)B_ * T_;

static constexpr size_t SZ_XBF   = (size_t)BT_ * IN_ * 2;
static constexpr size_t SZ_WIHE1 = 256 * 128 * 2;
static constexpr size_t SZ_WHHE1 = 256 * 64 * 2;
static constexpr size_t SZ_WIHE2 = 64 * 64 * 2;
static constexpr size_t SZ_WHHE2 = 64 * 32 * 2;  // K padded 16->32
static constexpr size_t SZ_WHHD1 = 256 * 64 * 2;
static constexpr size_t SZ_WIHD2 = 512 * 64 * 2;
static constexpr size_t SZ_WHHD2 = 512 * 128 * 2;
static constexpr size_t SZ_XGE1  = (size_t)BT_ * 256 * 4;
static constexpr size_t SZ_OUTE1 = (size_t)BT_ * 64 * 2;
static constexpr size_t SZ_XGE2  = (size_t)BT_ * 64 * 4;
static constexpr size_t SZ_OUTE2 = (size_t)BT_ * 16 * 2;
static constexpr size_t SZ_XGD1  = (size_t)B_ * 256 * 4;
static constexpr size_t SZ_OUTD1 = (size_t)BT_ * 64 * 2;

static constexpr size_t O_XBF   = 0;
static constexpr size_t O_WIHE1 = O_XBF + SZ_XBF;
static constexpr size_t O_WHHE1 = O_WIHE1 + SZ_WIHE1;
static constexpr size_t O_WIHE2 = O_WHHE1 + SZ_WHHE1;
static constexpr size_t O_WHHE2 = O_WIHE2 + SZ_WIHE2;
static constexpr size_t O_WHHD1 = O_WHHE2 + SZ_WHHE2;
static constexpr size_t O_WIHD2 = O_WHHD1 + SZ_WHHD1;
static constexpr size_t O_WHHD2 = O_WIHD2 + SZ_WIHD2;
static constexpr size_t O_XGE1  = O_WHHD2 + SZ_WHHD2;
static constexpr size_t O_OUTE1 = O_XGE1 + SZ_XGE1;
static constexpr size_t O_XGE2  = O_OUTE1 + SZ_OUTE1;
static constexpr size_t O_OUTE2 = O_XGE2 + SZ_XGE2;
static constexpr size_t O_XGD1  = O_OUTE2 + SZ_OUTE2;
static constexpr size_t O_OUTD1 = O_XGD1 + SZ_XGD1;
static constexpr size_t O_XGD2  = O_OUTD1 + SZ_OUTD1;

extern "C" void kernel_launch(void* const* d_in, const int* in_sizes, int n_in,
                              void* d_out, int out_size, void* d_ws,
                              size_t ws_size, hipStream_t stream) {
  const float* x      = (const float*)d_in[0];
  const float* Wih_e1 = (const float*)d_in[1];
  const float* Whh_e1 = (const float*)d_in[2];
  const float* bih_e1 = (const float*)d_in[3];
  const float* bhh_e1 = (const float*)d_in[4];
  const float* Wih_e2 = (const float*)d_in[5];
  const float* Whh_e2 = (const float*)d_in[6];
  const float* bih_e2 = (const float*)d_in[7];
  const float* bhh_e2 = (const float*)d_in[8];
  const float* Wih_d1 = (const float*)d_in[9];
  const float* Whh_d1 = (const float*)d_in[10];
  const float* bih_d1 = (const float*)d_in[11];
  const float* bhh_d1 = (const float*)d_in[12];
  const float* Wih_d2 = (const float*)d_in[13];
  const float* Whh_d2 = (const float*)d_in[14];
  const float* bih_d2 = (const float*)d_in[15];
  const float* bhh_d2 = (const float*)d_in[16];

  char* ws = (char*)d_ws;
  __bf16* xbf    = (__bf16*)(ws + O_XBF);
  __bf16* wih1b  = (__bf16*)(ws + O_WIHE1);
  __bf16* whh1b  = (__bf16*)(ws + O_WHHE1);
  __bf16* wih2b  = (__bf16*)(ws + O_WIHE2);
  __bf16* whh2b  = (__bf16*)(ws + O_WHHE2);
  __bf16* whhd1b = (__bf16*)(ws + O_WHHD1);
  __bf16* wihd2b = (__bf16*)(ws + O_WIHD2);
  __bf16* whhd2b = (__bf16*)(ws + O_WHHD2);
  float*  xg_e1  = (float*)(ws + O_XGE1);
  __bf16* out_e1 = (__bf16*)(ws + O_OUTE1);
  float*  xg_e2  = (float*)(ws + O_XGE2);
  __bf16* out_e2 = (__bf16*)(ws + O_OUTE2);
  float*  xg_d1  = (float*)(ws + O_XGD1);
  __bf16* out_d1 = (__bf16*)(ws + O_OUTD1);
  float*  xg_d2  = (float*)(ws + O_XGD2);
  float*  yout   = (float*)d_out;

  // --- conversions --------------------------------------------------------
  {
    long long n = BT_ * IN_;
    k_f32_to_bf16<<<dim3((unsigned)((n + 255) / 256)), dim3(256), 0, stream>>>(
        x, xbf, n);
  }
  auto cvt = [&](const float* s, __bf16* d, int N, int K, int KP) {
    int tot = N * KP;
    k_cvt_pad<<<dim3((tot + 255) / 256), dim3(256), 0, stream>>>(s, d, N, K,
                                                                 KP);
  };
  cvt(Wih_e1, wih1b, 256, 128, 128);
  cvt(Whh_e1, whh1b, 256, 64, 64);
  cvt(Wih_e2, wih2b, 64, 64, 64);
  cvt(Whh_e2, whh2b, 64, 16, 32);
  cvt(Whh_d1, whhd1b, 256, 64, 64);
  cvt(Wih_d2, wihd2b, 512, 64, 64);
  cvt(Whh_d2, whhd2b, 512, 128, 128);

  // --- e1: projection + scan ---------------------------------------------
  k_gemm_xg<<<dim3((unsigned)(BT_ / 64), 256 / 64), dim3(128), 0, stream>>>(
      xbf, wih1b, bih_e1, bhh_e1, xg_e1, 128, 256);
  k_lstm_scan<64, 8, true, __bf16><<<dim3(B_ / 16), dim3(256), 0, stream>>>(
      xg_e1, whh1b, out_e1, T_, (long long)T_ * 256, 256);

  // --- e2: projection + scan ---------------------------------------------
  k_gemm_xg<<<dim3((unsigned)(BT_ / 64), 64 / 64), dim3(128), 0, stream>>>(
      out_e1, wih2b, bih_e2, bhh_e2, xg_e2, 64, 64);
  k_lstm_scan<16, 4, true, __bf16><<<dim3(B_ / 16), dim3(128), 0, stream>>>(
      xg_e2, whh2b, out_e2, T_, (long long)T_ * 64, 64);

  // --- d1: latent projection (t-invariant xg) + scan ----------------------
  k_latent_xg<<<dim3(B_ * 256 / 256), dim3(256), 0, stream>>>(
      out_e2, Wih_d1, bih_d1, bhh_d1, xg_d1, T_);
  k_lstm_scan<64, 8, true, __bf16><<<dim3(B_ / 16), dim3(256), 0, stream>>>(
      xg_d1, whhd1b, out_d1, T_, 256, 0);

  // --- d2: projection + scan (fp32 out, no ReLU) --------------------------
  k_gemm_xg<<<dim3((unsigned)(BT_ / 64), 512 / 64), dim3(128), 0, stream>>>(
      out_d1, wihd2b, bih_d2, bhh_d2, xg_d2, 64, 512);
  k_lstm_scan<128, 16, false, float><<<dim3(B_ / 16), dim3(512), 0, stream>>>(
      xg_d2, whhd2b, yout, T_, (long long)T_ * 512, 512);
}
